// QWenBlock_77592879170073
// MI455X (gfx1250) — matrix operand order repaired and checked by reading, not compiled, not run on hardware
//
#include <hip/hip_runtime.h>
#include <hip/hip_bf16.h>

// ---------------------------------------------------------------------------
// Qwen block for MI455X (gfx1250, wave32, WMMA + TDM).
// fp32->bf16 once, all GEMMs via v_wmma_f32_16x16x32_bf16, flash attention,
// A-tile staging via tensor_load_to_lds (TDM, 6-arg toolchain variant).
// Requires ws_size >= ~800 MB.
// ---------------------------------------------------------------------------

#define SS 2048
#define DD 4096
#define HH 32
#define HDD 128
#define FFF 11008

typedef __bf16 bf16_t;
typedef __attribute__((ext_vector_type(16))) __bf16 v16bf;
typedef __attribute__((ext_vector_type(8)))  float  v8f;
typedef __attribute__((ext_vector_type(4)))  unsigned v4u;
typedef __attribute__((ext_vector_type(8)))  int      v8i;
typedef __attribute__((ext_vector_type(4)))  int      v4i;

#if defined(__has_builtin)
#  if __has_builtin(__builtin_amdgcn_tensor_load_to_lds)
#    define USE_TDM 1
#  endif
#endif
#ifndef USE_TDM
#  define USE_TDM 0
#endif

union Frag32B { uint4 u[2]; v16bf v; };
union Frag16B { uint4 u;    bf16_t h[8]; };

__device__ inline v8f v8f_zero() {
    v8f z;
#pragma unroll
    for (int i = 0; i < 8; ++i) z[i] = 0.0f;
    return z;
}

// ---------------------------------------------------------------------------
// fp32 -> bf16 conversion (grid-stride, bandwidth bound)
// ---------------------------------------------------------------------------
__global__ void cvt_f32_bf16(const float* __restrict__ in, bf16_t* __restrict__ out, int n) {
    int i = blockIdx.x * blockDim.x + threadIdx.x;
    int stride = gridDim.x * blockDim.x;
    for (; i < n; i += stride) out[i] = (bf16_t)in[i];
}

// ---------------------------------------------------------------------------
// RMSNorm: one block per row of [SS, DD]; wave32 shuffle + LDS reduction.
// ---------------------------------------------------------------------------
__global__ __launch_bounds__(256) void rmsnorm_bf16(const float* __restrict__ x,
                                                    const float* __restrict__ g,
                                                    bf16_t* __restrict__ out) {
    int row = blockIdx.x;
    const float* xr = x + (size_t)row * DD;
    float ss = 0.0f;
    for (int i = threadIdx.x; i < DD; i += blockDim.x) {
        float v = xr[i];
        ss += v * v;
    }
    __shared__ float red[32];
#pragma unroll
    for (int off = 16; off > 0; off >>= 1) ss += __shfl_down(ss, off, 32);
    int wid = threadIdx.x >> 5, lid = threadIdx.x & 31;
    if (lid == 0) red[wid] = ss;
    __syncthreads();
    if (wid == 0) {
        float v = (lid < (int)(blockDim.x >> 5)) ? red[lid] : 0.0f;
#pragma unroll
        for (int off = 16; off > 0; off >>= 1) v += __shfl_down(v, off, 32);
        if (lid == 0) red[0] = rsqrtf(v / (float)DD + 1e-6f);
    }
    __syncthreads();
    float inv = red[0];
    bf16_t* orow = out + (size_t)row * DD;
    for (int i = threadIdx.x; i < DD; i += blockDim.x)
        orow[i] = (bf16_t)(xr[i] * inv * g[i]);
}

// ---------------------------------------------------------------------------
// bf16 WMMA GEMM: C[M,N](f32) = A[M,K](bf16,row) * B[K,N](bf16,row)
// Block tile 128x128, BK=32, 256 threads = 8 waves, wave tile 32x64 (2x4 WMMA).
// A tile: DMA'd whole by ONE tensor_load_to_lds (D# pad fields produce the
// padded 40-bf16 LDS row stride); B tile: staged transposed by all threads
// (overlaps the TDM). Fragments follow ISA 7.12.2 per-lane layouts.
// Requires M%128==0, N%128==0, K%32==0 (true here).
// ---------------------------------------------------------------------------
#define BM 128
#define BN 128
#define BK 32
#define LPAD 8

__global__ __launch_bounds__(256) void gemm_bf16_wmma(const bf16_t* __restrict__ A,
                                                      const bf16_t* __restrict__ Bm,
                                                      float* __restrict__ C,
                                                      int M, int N, int K) {
    __shared__ bf16_t As[BM][BK + LPAD];  // 128*40*2 = 10.0 KB
    __shared__ bf16_t Bs[BN][BK + LPAD];  // 128*40*2 = 10.0 KB (transposed: Bs[n][k])

    const int tid   = threadIdx.x;
    const int lane  = tid & 31;
    const int wid   = tid >> 5;
    const int lhalf = lane >> 4;    // 0/1
    const int lmod  = lane & 15;
    const int wave_m = wid & 3;     // 0..3 -> 4 * 32 = 128 rows
    const int wave_n = wid >> 2;    // 0..1 -> 2 * 64 = 128 cols

    const int row0 = blockIdx.y * BM;
    const int col0 = blockIdx.x * BN;

    v8f acc[2][4];
#pragma unroll
    for (int mi = 0; mi < 2; ++mi)
#pragma unroll
        for (int ni = 0; ni < 4; ++ni) acc[mi][ni] = v8f_zero();

    // global->LDS mapping
    const int a_row = tid >> 1;          // 0..127
    const int a_cc  = (tid & 1) * 16;    // 0 / 16
    const int b_k   = tid >> 3;          // 0..31
    const int b_n0  = (tid & 7) * 16;    // 0..112

#if USE_TDM
    // TDM descriptor constants (group1): data_size=2B, pad_enable,
    // pad_interval=16 DWORDs (code 3), pad_amount=4 DWORDs (code 3);
    // tensor = [M rows x K cols], tile = 128 x 32.
    const unsigned g1w0 = (1u << 16) | (1u << 20) | (3u << 22) | (3u << 25);
#endif

    for (int k0 = 0; k0 < K; k0 += BK) {
#if USE_TDM
        if (wid == 0) {
            unsigned lds_base = (unsigned)(size_t)(&As[0][0]);
            unsigned long long ga =
                (unsigned long long)(size_t)(A + (size_t)row0 * K + k0);
            v4u g0;
            g0[0] = 1u;                                   // count=1 (valid user D#)
            g0[1] = lds_base;                             // lds_addr
            g0[2] = (unsigned)(ga & 0xFFFFFFFFu);         // global_addr[31:0]
            g0[3] = (unsigned)((ga >> 32) & 0x1FFFFFFu)   // global_addr[56:32]
                    | 0x80000000u;                        // type=2 ("image")
            v8i g1;
            g1[0] = (int)g1w0;
            g1[1] = (int)(((unsigned)K & 0xFFFFu) << 16);               // tensor_dim0 lo
            g1[2] = (int)((((unsigned)K >> 16) & 0xFFFFu)               // tensor_dim0 hi
                          | (((unsigned)M & 0xFFFFu) << 16));           // tensor_dim1 lo
            g1[3] = (int)(32u << 16);                                   // tile_dim0 = 32
            g1[4] = 128;                                                // tile_dim1 = 128
            g1[5] = (int)(unsigned)K;                                   // dim0 stride lo
            g1[6] = 0;
            g1[7] = 0;
            v4i gz4;
            gz4[0] = gz4[1] = gz4[2] = gz4[3] = 0;
            v8i gz8;
#pragma unroll
            for (int i = 0; i < 8; ++i) gz8[i] = 0;
            __builtin_amdgcn_tensor_load_to_lds(g0, g1, gz4, gz4, gz8, 0);
        }
#else
        // stage A tile (each thread: 16 bf16 = two 16B loads)
        {
            const uint4* ap = reinterpret_cast<const uint4*>(
                A + (size_t)(row0 + a_row) * K + k0 + a_cc);
            uint4 v0 = ap[0], v1 = ap[1];
            uint4* dst = reinterpret_cast<uint4*>(&As[a_row][a_cc]);
            dst[0] = v0; dst[1] = v1;
        }
#endif
        // stage B tile transposed (each thread: 16 bf16 along N, scatter by k)
        {
            const uint4* bp = reinterpret_cast<const uint4*>(
                Bm + (size_t)(k0 + b_k) * N + col0 + b_n0);
            Frag16B b0, b1;
            b0.u = bp[0];
            b1.u = bp[1];
#pragma unroll
            for (int i = 0; i < 8; ++i) Bs[b_n0 + i][b_k] = b0.h[i];
#pragma unroll
            for (int i = 0; i < 8; ++i) Bs[b_n0 + 8 + i][b_k] = b1.h[i];
        }
#if USE_TDM
        if (wid == 0) __builtin_amdgcn_s_wait_tensorcnt(0);
#endif
        __syncthreads();

        // fragments per ISA layout
        Frag32B af[2], bf[4];
#pragma unroll
        for (int mi = 0; mi < 2; ++mi) {
            int r = wave_m * 32 + mi * 16 + lmod;
            int base = 8 * lhalf;
            af[mi].u[0] = *reinterpret_cast<const uint4*>(&As[r][base]);
            af[mi].u[1] = *reinterpret_cast<const uint4*>(&As[r][base + 16]);
        }
#pragma unroll
        for (int ni = 0; ni < 4; ++ni) {
            int n = wave_n * 64 + ni * 16 + lmod;
            int klo = 8 * lhalf;
            bf[ni].u[0] = *reinterpret_cast<const uint4*>(&Bs[n][klo]);
            bf[ni].u[1] = *reinterpret_cast<const uint4*>(&Bs[n][klo + 16]);
        }
#pragma unroll
        for (int mi = 0; mi < 2; ++mi)
#pragma unroll
            for (int ni = 0; ni < 4; ++ni)
                acc[mi][ni] = __builtin_amdgcn_wmma_f32_16x16x32_bf16(
                    false, af[mi].v, false, bf[ni].v, (short)0, acc[mi][ni],
                    false, false);
        __syncthreads();
    }

    // epilogue: C-tile layout -> global (lane = N, vgpr index = M row)
#pragma unroll
    for (int mi = 0; mi < 2; ++mi)
#pragma unroll
        for (int ni = 0; ni < 4; ++ni) {
            int colg = col0 + wave_n * 64 + ni * 16 + lmod;
#pragma unroll
            for (int v = 0; v < 8; ++v) {
                int rowg = row0 + wave_m * 32 + mi * 16 + v + 8 * lhalf;
                C[(size_t)rowg * N + colg] = acc[mi][ni][v];
            }
        }
}

// ---------------------------------------------------------------------------
// QKV epilogue: bias + RoPE, split into q,k [H,S,HD] bf16 and v^T [H,HD,S] bf16
// ---------------------------------------------------------------------------
__global__ void qkv_post(const float* __restrict__ qkv, const float* __restrict__ bias,
                         const float* __restrict__ cosT, const float* __restrict__ sinT,
                         bf16_t* __restrict__ q, bf16_t* __restrict__ k,
                         bf16_t* __restrict__ vT) {
    int idx = blockIdx.x * blockDim.x + threadIdx.x;
    if (idx >= SS * DD) return;
    int s = idx / DD, d = idx % DD;
    int h = d / HDD, hd = d % HDD;
    const float* row = qkv + (size_t)s * 3 * DD;
    float qv = row[d] + bias[d];
    float kv = row[DD + d] + bias[DD + d];
    float vv = row[2 * DD + d] + bias[2 * DD + d];
    int hd2 = (hd < HDD / 2) ? hd + HDD / 2 : hd - HDD / 2;
    float sgn = (hd < HDD / 2) ? -1.0f : 1.0f;
    int d2 = h * HDD + hd2;
    float q2 = row[d2] + bias[d2];
    float k2 = row[DD + d2] + bias[DD + d2];
    float c = cosT[s * HDD + hd], sn = sinT[s * HDD + hd];
    float qo = qv * c + sgn * q2 * sn;
    float ko = kv * c + sgn * k2 * sn;
    size_t o = ((size_t)h * SS + s) * HDD + hd;
    q[o] = (bf16_t)qo;
    k[o] = (bf16_t)ko;
    vT[((size_t)h * HDD + hd) * SS + s] = (bf16_t)vv;
}

// ---------------------------------------------------------------------------
// Flash attention (causal), all-WMMA.
// grid = (S/128, H); 256 thr = 8 waves; each wave owns 16 query rows.
// ---------------------------------------------------------------------------
__global__ __launch_bounds__(256) void flash_attn(const bf16_t* __restrict__ q,
                                                  const bf16_t* __restrict__ k,
                                                  const bf16_t* __restrict__ vT,
                                                  bf16_t* __restrict__ ctx) {
    __shared__ bf16_t Ps[8][16][32 + LPAD];  // 10 KB, wave-private slabs

    const int tid   = threadIdx.x;
    const int lane  = tid & 31;
    const int wid   = tid >> 5;
    const int lhalf = lane >> 4;
    const int lmod  = lane & 15;

    const int h  = blockIdx.y;
    const int q0 = blockIdx.x * 128 + wid * 16;

    const bf16_t* qh = q  + (size_t)h * SS * HDD;
    const bf16_t* kh = k  + (size_t)h * SS * HDD;
    const bf16_t* vh = vT + (size_t)h * HDD * SS;

    Frag32B qf[4];
    {
        int qrow = q0 + lmod;
#pragma unroll
        for (int c = 0; c < 4; ++c) {
            int base = c * 32 + 8 * lhalf;
            const bf16_t* p = qh + (size_t)qrow * HDD + base;
            qf[c].u[0] = *reinterpret_cast<const uint4*>(p);
            qf[c].u[1] = *reinterpret_cast<const uint4*>(p + 16);
        }
    }

    v8f acc[8];
#pragma unroll
    for (int t = 0; t < 8; ++t) acc[t] = v8f_zero();
    float mrow[8], lrow[8];
#pragma unroll
    for (int v = 0; v < 8; ++v) { mrow[v] = -1e30f; lrow[v] = 0.0f; }

    const float scale = 0.08838834764831845f;  // 1/sqrt(128)

    for (int j0 = 0; j0 < q0 + 16; j0 += 32) {
        v8f s[2];
#pragma unroll
        for (int nt = 0; nt < 2; ++nt) {
            v8f sv = v8f_zero();
            int key = j0 + nt * 16 + lmod;
#pragma unroll
            for (int c = 0; c < 4; ++c) {
                Frag32B kf;
                int base = c * 32 + 8 * lhalf;
                const bf16_t* p = kh + (size_t)key * HDD + base;
                kf.u[0] = *reinterpret_cast<const uint4*>(p);
                kf.u[1] = *reinterpret_cast<const uint4*>(p + 16);
                sv = __builtin_amdgcn_wmma_f32_16x16x32_bf16(
                    false, qf[c].v, false, kf.v, (short)0, sv, false, false);
            }
            s[nt] = sv;
        }
#pragma unroll
        for (int nt = 0; nt < 2; ++nt) {
            int key = j0 + nt * 16 + lmod;
#pragma unroll
            for (int v = 0; v < 8; ++v) {
                int qi = q0 + v + 8 * lhalf;
                float val = s[nt][v] * scale;
                s[nt][v] = (key <= qi) ? val : -1e30f;
            }
        }
#pragma unroll
        for (int v = 0; v < 8; ++v) {
            float rmax = fmaxf(s[0][v], s[1][v]);
#pragma unroll
            for (int off = 8; off > 0; off >>= 1)
                rmax = fmaxf(rmax, __shfl_xor(rmax, off, 32));
            float mnew = fmaxf(mrow[v], rmax);
            float corr = __expf(mrow[v] - mnew);
            float p0 = __expf(s[0][v] - mnew);
            float p1 = __expf(s[1][v] - mnew);
            s[0][v] = p0; s[1][v] = p1;
            float rsum = p0 + p1;
#pragma unroll
            for (int off = 8; off > 0; off >>= 1)
                rsum += __shfl_xor(rsum, off, 32);
            lrow[v] = lrow[v] * corr + rsum;
            mrow[v] = mnew;
#pragma unroll
            for (int t = 0; t < 8; ++t) acc[t][v] *= corr;
        }
        // P: C-layout -> A-layout via wave-private LDS (in-order DS + wait)
#pragma unroll
        for (int nt = 0; nt < 2; ++nt)
#pragma unroll
            for (int v = 0; v < 8; ++v)
                Ps[wid][v + 8 * lhalf][nt * 16 + lmod] = (bf16_t)s[nt][v];
        asm volatile("s_wait_dscnt 0" ::: "memory");
        Frag32B pf;
        {
            int base = 8 * lhalf;
            pf.u[0] = *reinterpret_cast<const uint4*>(&Ps[wid][lmod][base]);
            pf.u[1] = *reinterpret_cast<const uint4*>(&Ps[wid][lmod][base + 16]);
        }
#pragma unroll
        for (int t = 0; t < 8; ++t) {
            Frag32B vf;
            int hd = t * 16 + lmod;
            const bf16_t* p = vh + (size_t)hd * SS + j0 + 8 * lhalf;
            vf.u[0] = *reinterpret_cast<const uint4*>(p);
            vf.u[1] = *reinterpret_cast<const uint4*>(p + 16);
            acc[t] = __builtin_amdgcn_wmma_f32_16x16x32_bf16(
                false, pf.v, false, vf.v, (short)0, acc[t], false, false);
        }
    }

#pragma unroll
    for (int t = 0; t < 8; ++t) {
        int hd = t * 16 + lmod;
#pragma unroll
        for (int v = 0; v < 8; ++v) {
            int qi = q0 + v + 8 * lhalf;
            ctx[(size_t)qi * DD + h * HDD + hd] = (bf16_t)(acc[t][v] / lrow[v]);
        }
    }
}

// ---------------------------------------------------------------------------
// elementwise helpers
// ---------------------------------------------------------------------------
__global__ void add_f32(const float* __restrict__ a, const float* __restrict__ b,
                        float* __restrict__ o, int n) {
    int i = blockIdx.x * blockDim.x + threadIdx.x;
    int stride = gridDim.x * blockDim.x;
    for (; i < n; i += stride) o[i] = a[i] + b[i];
}

__global__ void swiglu_bf16(const float* __restrict__ a1, const float* __restrict__ a2,
                            bf16_t* __restrict__ o, int n) {
    int i = blockIdx.x * blockDim.x + threadIdx.x;
    int stride = gridDim.x * blockDim.x;
    for (; i < n; i += stride) {
        float x = a2[i];
        float silu = x / (1.0f + __expf(-x));
        o[i] = (bf16_t)(a1[i] * silu);
    }
}

// ---------------------------------------------------------------------------
// launch
// ---------------------------------------------------------------------------
extern "C" void kernel_launch(void* const* d_in, const int* in_sizes, int n_in,
                              void* d_out, int out_size, void* d_ws, size_t ws_size,
                              hipStream_t stream) {
    const float* h_in  = (const float*)d_in[0];   // [S,D]
    const float* cosT  = (const float*)d_in[1];   // [S,HD]
    const float* sinT  = (const float*)d_in[2];   // [S,HD]
    // d_in[3] attn_mask: pure causal, applied in-kernel
    const float* g1    = (const float*)d_in[4];
    const float* g2    = (const float*)d_in[5];
    const float* Wqkv  = (const float*)d_in[6];   // [D,3D]
    const float* bqkv  = (const float*)d_in[7];   // [3D]
    const float* Wo    = (const float*)d_in[8];   // [D,D]
    const float* w1    = (const float*)d_in[9];   // [D,FF]
    const float* w2    = (const float*)d_in[10];  // [D,FF]
    const float* wc    = (const float*)d_in[11];  // [FF,D]
    float* out = (float*)d_out;                   // [S,D]

    char* ws = (char*)d_ws;
    size_t o = 0;
    bf16_t* wqkv_b = (bf16_t*)(ws + o); o += (size_t)DD * 3 * DD * 2;
    bf16_t* wo_b   = (bf16_t*)(ws + o); o += (size_t)DD * DD * 2;
    bf16_t* w1_b   = (bf16_t*)(ws + o); o += (size_t)DD * FFF * 2;
    bf16_t* w2_b   = (bf16_t*)(ws + o); o += (size_t)DD * FFF * 2;
    bf16_t* wc_b   = (bf16_t*)(ws + o); o += (size_t)FFF * DD * 2;
    bf16_t* xb     = (bf16_t*)(ws + o); o += (size_t)SS * DD * 2;
    float*  qkv_f  = (float*) (ws + o); o += (size_t)SS * 3 * DD * 4;   // reused as a1
    bf16_t* qb     = (bf16_t*)(ws + o); o += (size_t)SS * DD * 2;
    bf16_t* kb     = (bf16_t*)(ws + o); o += (size_t)SS * DD * 2;
    bf16_t* vtb    = (bf16_t*)(ws + o); o += (size_t)SS * DD * 2;
    bf16_t* ctxb   = (bf16_t*)(ws + o); o += (size_t)SS * DD * 2;
    float*  attn_f = (float*) (ws + o); o += (size_t)SS * DD * 4;       // reused as mlp
    float*  li_f   = (float*) (ws + o); o += (size_t)SS * DD * 4;
    float*  a2_f   = (float*) (ws + o); o += (size_t)SS * FFF * 4;
    bf16_t* mb     = (bf16_t*)(ws + o); o += (size_t)SS * FFF * 2;
    float*  a1_f   = qkv_f;
    float*  mlp_f  = attn_f;

    const int T = 256;
    cvt_f32_bf16<<<2048, T, 0, stream>>>(Wqkv, wqkv_b, DD * 3 * DD);
    cvt_f32_bf16<<<2048, T, 0, stream>>>(Wo,   wo_b,   DD * DD);
    cvt_f32_bf16<<<2048, T, 0, stream>>>(w1,   w1_b,   DD * FFF);
    cvt_f32_bf16<<<2048, T, 0, stream>>>(w2,   w2_b,   DD * FFF);
    cvt_f32_bf16<<<2048, T, 0, stream>>>(wc,   wc_b,   FFF * DD);

    rmsnorm_bf16<<<SS, T, 0, stream>>>(h_in, g1, xb);

    gemm_bf16_wmma<<<dim3(3 * DD / BN, SS / BM), T, 0, stream>>>(xb, wqkv_b, qkv_f, SS, 3 * DD, DD);
    qkv_post<<<(SS * DD + T - 1) / T, T, 0, stream>>>(qkv_f, bqkv, cosT, sinT, qb, kb, vtb);
    flash_attn<<<dim3(SS / 128, HH), T, 0, stream>>>(qb, kb, vtb, ctxb);

    gemm_bf16_wmma<<<dim3(DD / BN, SS / BM), T, 0, stream>>>(ctxb, wo_b, attn_f, SS, DD, DD);
    add_f32<<<2048, T, 0, stream>>>(h_in, attn_f, li_f, SS * DD);

    rmsnorm_bf16<<<SS, T, 0, stream>>>(li_f, g2, xb);
    gemm_bf16_wmma<<<dim3(FFF / BN, SS / BM), T, 0, stream>>>(xb, w1_b, a1_f, SS, FFF, DD);
    gemm_bf16_wmma<<<dim3(FFF / BN, SS / BM), T, 0, stream>>>(xb, w2_b, a2_f, SS, FFF, DD);
    swiglu_bf16<<<2048, T, 0, stream>>>(a1_f, a2_f, mb, SS * FFF);
    gemm_bf16_wmma<<<dim3(DD / BN, SS / BM), T, 0, stream>>>(mb, wc_b, mlp_f, SS, DD, FFF);

    add_f32<<<2048, T, 0, stream>>>(li_f, mlp_f, out, SS * DD);
}
